// DuelingNetWork_59614146068858
// MI455X (gfx1250) — matrix operand order repaired and checked by reading, not compiled, not run on hardware
//
#include <hip/hip_runtime.h>
#include <math.h>

typedef __attribute__((ext_vector_type(16))) _Float16 v16h;
typedef __attribute__((ext_vector_type(8)))  float    v8f;

// ---- problem sizes ----
// B=8, D=64, O=256, I=256, HID=128
// ws layout (float offsets):
//   S2 : [0, 1024)                 8 x 128
//   U2 : [1024, 1024+262144)       2048 x 128   rows = (b*256 + o)
//   V2 : [263168, 263168+262144)   2048 x 128   rows = (b*256 + i)
//   Wc (f16) at byte 2101248: Wc_o (128x64) then Wc_i (128x64)
#define S2_OFF   0
#define U2_OFF   1024
#define V2_OFF   263168
#define WC_BYTE_OFF 2101248u

// A output at out+8; hn at out+524296; cn at out+524808
#define A_OFF    8
#define HN_OFF   524296
#define CN_OFF   524808

// ---------------------------------------------------------------------------
// Kernel 1: state mean, LSTM gates, hn/cn, V head, and S2 = Wa1@(Ws@hn+ba0)+ba1
// All small (sub-MFLOP), single block.
// ---------------------------------------------------------------------------
__global__ __launch_bounds__(256) void k1_small(
    const float* __restrict__ ope_gat, const float* __restrict__ h_in,
    const float* __restrict__ c_in,
    const float* __restrict__ W_ih, const float* __restrict__ W_hh,
    const float* __restrict__ b_ih, const float* __restrict__ b_hh,
    const float* __restrict__ Wv0, const float* __restrict__ bv0,
    const float* __restrict__ Wv1, const float* __restrict__ bv1,
    const float* __restrict__ Wv2, const float* __restrict__ bv2,
    const float* __restrict__ Wa0, const float* __restrict__ ba0,
    const float* __restrict__ Wa1, const float* __restrict__ ba1,
    float* __restrict__ out, float* __restrict__ wsf)
{
  __shared__ float state[512];   // 8x64
  __shared__ float gates[2048];  // 8x256
  __shared__ float hn_s[512];    // 8x64
  __shared__ float z0[1024];     // 8x128
  __shared__ float z1[1024];     // 8x128
  __shared__ float sb[1024];     // 8x128
  const int t = threadIdx.x;

  // state[b][d] = mean over o of ope_gat[b][o+1][d]
  for (int k = t; k < 512; k += 256) {
    int b = k >> 6, d = k & 63;
    const float* p = ope_gat + (b*258 + 1)*64 + d;
    float s = 0.f;
    for (int o = 0; o < 256; ++o) s += p[o*64];
    state[k] = s * (1.0f/256.0f);
  }
  __syncthreads();

  // gates = state@W_ih.T + b_ih + h@W_hh.T + b_hh
  for (int k = t; k < 2048; k += 256) {
    int b = k >> 8, j = k & 255;
    float acc = b_ih[j] + b_hh[j];
    const float* wi = W_ih + j*64;
    const float* wh = W_hh + j*64;
    const float* st = state + b*64;
    const float* hv = h_in + b*64;
    for (int d = 0; d < 64; ++d) acc = fmaf(st[d], wi[d], fmaf(hv[d], wh[d], acc));
    gates[k] = acc;
  }
  __syncthreads();

  // LSTM cell (split order: i, f, g, o)
  for (int k = t; k < 512; k += 256) {
    int b = k >> 6, d = k & 63;
    float ig = gates[b*256 + d];
    float fg = gates[b*256 + 64 + d];
    float gg = gates[b*256 + 128 + d];
    float og = gates[b*256 + 192 + d];
    float si = 1.f/(1.f+expf(-ig));
    float sf = 1.f/(1.f+expf(-fg));
    float so = 1.f/(1.f+expf(-og));
    float cn = sf*c_in[k] + si*tanhf(gg);
    float hn = so*tanhf(cn);
    out[HN_OFF + k] = hn;
    out[CN_OFF + k] = cn;
    hn_s[k] = hn;
  }
  __syncthreads();

  // V head: z0 = hn@Wv0.T + bv0  (NO activation per reference)
  for (int k = t; k < 1024; k += 256) {
    int b = k >> 7, j = k & 127;
    float acc = bv0[j];
    const float* w = Wv0 + j*64;
    const float* hv = hn_s + b*64;
    for (int d = 0; d < 64; ++d) acc = fmaf(hv[d], w[d], acc);
    z0[k] = acc;
  }
  __syncthreads();
  // z1 = lrelu(z0@Wv1.T + bv1)
  for (int k = t; k < 1024; k += 256) {
    int b = k >> 7, j = k & 127;
    float acc = bv1[j];
    const float* w = Wv1 + j*128;
    const float* zz = z0 + b*128;
    for (int d = 0; d < 128; ++d) acc = fmaf(zz[d], w[d], acc);
    z1[k] = fmaxf(acc, 0.01f*acc);
  }
  __syncthreads();
  // V = z1@Wv2.T + bv2
  if (t < 8) {
    float acc = bv2[0];
    const float* zz = z1 + t*128;
    for (int d = 0; d < 128; ++d) acc = fmaf(zz[d], Wv2[d], acc);
    out[t] = acc;
  }
  // sb[b][g] = hn@Ws.T + ba0   (Ws = Wa0[:, :64], row stride 192)
  for (int k = t; k < 1024; k += 256) {
    int b = k >> 7, g = k & 127;
    float acc = ba0[g];
    const float* w = Wa0 + g*192;
    const float* hv = hn_s + b*64;
    for (int d = 0; d < 64; ++d) acc = fmaf(hv[d], w[d], acc);
    sb[k] = acc;
  }
  __syncthreads();
  // S2[b][h] = Wa1@sb + ba1  -> ws
  for (int k = t; k < 1024; k += 256) {
    int b = k >> 7, hh = k & 127;
    float acc = ba1[hh];
    const float* w = Wa1 + hh*128;
    const float* sv = sb + b*128;
    for (int g = 0; g < 128; ++g) acc = fmaf(sv[g], w[g], acc);
    wsf[S2_OFF + k] = acc;
  }
}

// ---------------------------------------------------------------------------
// Kernel 2: fused weights  Wc_o = Wa1 @ Wa0[:,64:128],  Wc_i = Wa1 @ Wa0[:,128:192]
// stored as f16 (128x64 each), WMMA-B-fragment friendly (row-major, 32B aligned rows)
// ---------------------------------------------------------------------------
__global__ __launch_bounds__(256) void k2_wc(
    const float* __restrict__ Wa0, const float* __restrict__ Wa1,
    _Float16* __restrict__ wc)
{
  int idx = blockIdx.x*256 + threadIdx.x;   // 0..16383
  int which = idx >> 13;                    // 0 -> Wc_o, 1 -> Wc_i
  int r = idx & 8191;
  int g = r >> 6;
  int d = r & 63;
  int col = 64 + which*64 + d;
  const float* wrow = Wa1 + g*128;
  float acc = 0.f;
  for (int h = 0; h < 128; ++h) acc = fmaf(wrow[h], Wa0[h*192 + col], acc);
  wc[which*8192 + g*64 + d] = (_Float16)acc;
}

// ---------------------------------------------------------------------------
// Kernel 3: WMMA GEMMs
//   U2 (2048x128) = ope_d      (2048x64) @ Wc_o.T (64x128)
//   V2 (2048x128) = insert_gat (2048x64) @ Wc_i.T (64x128)
// One wave per 16-row M-tile; N=128 in 8 tiles; K=64 in 2 chunks of 32.
// v_wmma_f32_16x16x32_f16, f32 accumulate.
// ---------------------------------------------------------------------------
__device__ inline v16h load_a_frag(const float* __restrict__ arow, int kc, int half_id)
{
  // A frag, contiguous k order (B uses the same): lane m=L%16, half=L/16; element e -> K = half*16 + e + kc*32
  int base = kc*32 + half_id*16;
  v16h av;
  #pragma unroll
  for (int e = 0; e < 8; ++e)  av[e]   = (_Float16)arow[base + e];
  #pragma unroll
  for (int e = 0; e < 8; ++e)  av[8+e] = (_Float16)arow[base + 8 + e];
  return av;
}

__global__ __launch_bounds__(32) void k3_wmma(
    const float* __restrict__ ope_gat, const float* __restrict__ insert_gat,
    const _Float16* __restrict__ wc, float* __restrict__ wsf)
{
  const int tile = blockIdx.x;      // 0..255
  const int which = tile >> 7;      // 0 = U2 (ope), 1 = V2 (insert)
  const int mt = tile & 127;
  const int lane = threadIdx.x;
  const int m = lane & 15;
  const int half_id = lane >> 4;

  const int R = mt*16 + m;          // global row for this lane's A rows
  const int b = R >> 8;
  const int o = R & 255;
  const float* arow = which ? (insert_gat + (b*256 + o)*64)
                            : (ope_gat   + (b*258 + o + 1)*64);  // ope = ope_gat[:,1:-1]

  const v16h a0 = load_a_frag(arow, 0, half_id);
  const v16h a1 = load_a_frag(arow, 1, half_id);

  const _Float16* wcb = wc + which*8192;          // 128x64 f16, row-major
  float* dstbase = wsf + (which ? V2_OFF : U2_OFF) + (mt*16)*128;
  const int n = lane & 15;
  const int kc2 = lane >> 4;        // B K-half / D row-half
  const int rbase = 8*kc2;

  #pragma unroll
  for (int nt = 0; nt < 8; ++nt) {
    const _Float16* brow = wcb + (nt*16 + n)*64;   // B[k][n] = Wc[g][d], g = nt*16+n
    v16h b0 = *(const v16h*)(brow + kc2*16);       // K = 0..31
    v16h b1 = *(const v16h*)(brow + 32 + kc2*16);  // K = 32..63
    v8f acc = {};
    acc = __builtin_amdgcn_wmma_f32_16x16x32_f16(false, a0, false, b0,
                                                 (short)0, acc, false, false);
    acc = __builtin_amdgcn_wmma_f32_16x16x32_f16(false, a1, false, b1,
                                                 (short)0, acc, false, false);
    float* dst = dstbase + nt*16 + n;
    #pragma unroll
    for (int r = 0; r < 8; ++r)
      dst[(rbase + r)*128] = acc[r];               // D row = r + 8*(lane/16)
  }
}

// ---------------------------------------------------------------------------
// Kernel 4: A[b, o*256+i] = ba2 + sum_h Wa2[h] * lrelu(S2[b,h]+U2[b,o,h]+V2[b,i,h])
// Block = (b, 8 o's, 64 i's). V2 chunk + per-o fused (S2+U2) rows staged in LDS.
// 132-float padded V2 rows -> conflict-free ds_load_b128 across lanes.
// ---------------------------------------------------------------------------
__global__ __launch_bounds__(256) void k4_attn(
    const float* __restrict__ wsf, const float* __restrict__ Wa2,
    const float* __restrict__ ba2, float* __restrict__ out)
{
  __shared__ float V2s[64*132];
  __shared__ float w_s[8*128];
  __shared__ float wa2_s[128];

  const int i0 = blockIdx.x * 64;
  const int o0 = blockIdx.y * 8;
  const int b  = blockIdx.z;
  const int t  = threadIdx.x;

  const float* V2g = wsf + V2_OFF + (b*256 + i0)*128;
  for (int k = t; k < 64*128; k += 256) {
    int ii = k >> 7, hh = k & 127;
    V2s[ii*132 + hh] = V2g[k];
  }
  const float* S2g = wsf + S2_OFF + b*128;
  const float* U2g = wsf + U2_OFF + (b*256 + o0)*128;
  for (int k = t; k < 1024; k += 256) {
    int wv = k >> 7, hh = k & 127;
    w_s[k] = S2g[hh] + U2g[wv*128 + hh];
  }
  if (t < 128) wa2_s[t] = Wa2[t];
  __syncthreads();

  const int wave = t >> 5, lane = t & 31;
  const int o = o0 + wave;
  const float* wrow = w_s + wave*128;
  const int iA = lane*2, iB = lane*2 + 1;
  const float* vA = V2s + iA*132;
  const float* vB = V2s + iB*132;
  float accA = ba2[0], accB = ba2[0];

  #pragma unroll 8
  for (int h = 0; h < 128; h += 4) {
    float4 w4 = *(const float4*)(wrow + h);   // LDS broadcast
    float4 s4 = *(const float4*)(wa2_s + h);  // LDS broadcast
    float4 a4 = *(const float4*)(vA + h);
    float4 b4 = *(const float4*)(vB + h);
    float x;
    x = w4.x + a4.x; x = fmaxf(x, 0.01f*x); accA = fmaf(x, s4.x, accA);
    x = w4.x + b4.x; x = fmaxf(x, 0.01f*x); accB = fmaf(x, s4.x, accB);
    x = w4.y + a4.y; x = fmaxf(x, 0.01f*x); accA = fmaf(x, s4.y, accA);
    x = w4.y + b4.y; x = fmaxf(x, 0.01f*x); accB = fmaf(x, s4.y, accB);
    x = w4.z + a4.z; x = fmaxf(x, 0.01f*x); accA = fmaf(x, s4.z, accA);
    x = w4.z + b4.z; x = fmaxf(x, 0.01f*x); accB = fmaf(x, s4.z, accB);
    x = w4.w + a4.w; x = fmaxf(x, 0.01f*x); accA = fmaf(x, s4.w, accA);
    x = w4.w + b4.w; x = fmaxf(x, 0.01f*x); accB = fmaf(x, s4.w, accB);
  }

  float* Arow = out + A_OFF + (size_t)b*65536 + o*256 + i0;
  Arow[iA] = accA;
  Arow[iB] = accB;
}

// ---------------------------------------------------------------------------
extern "C" void kernel_launch(void* const* d_in, const int* in_sizes, int n_in,
                              void* d_out, int out_size, void* d_ws, size_t ws_size,
                              hipStream_t stream) {
  (void)in_sizes; (void)n_in; (void)out_size; (void)ws_size;
  const float* ope_gat    = (const float*)d_in[0];
  const float* insert_gat = (const float*)d_in[1];
  const float* h_in = (const float*)d_in[2];
  const float* c_in = (const float*)d_in[3];
  const float* W_ih = (const float*)d_in[4];
  const float* W_hh = (const float*)d_in[5];
  const float* b_ih = (const float*)d_in[6];
  const float* b_hh = (const float*)d_in[7];
  const float* Wv0  = (const float*)d_in[8];
  const float* bv0  = (const float*)d_in[9];
  const float* Wv1  = (const float*)d_in[10];
  const float* bv1  = (const float*)d_in[11];
  const float* Wv2  = (const float*)d_in[12];
  const float* bv2  = (const float*)d_in[13];
  const float* Wa0  = (const float*)d_in[14];
  const float* ba0  = (const float*)d_in[15];
  const float* Wa1  = (const float*)d_in[16];
  const float* ba1  = (const float*)d_in[17];
  const float* Wa2  = (const float*)d_in[18];
  const float* ba2  = (const float*)d_in[19];

  float* out = (float*)d_out;
  float* wsf = (float*)d_ws;
  _Float16* wc = (_Float16*)((char*)d_ws + WC_BYTE_OFF);

  k1_small<<<1, 256, 0, stream>>>(ope_gat, h_in, c_in, W_ih, W_hh, b_ih, b_hh,
                                  Wv0, bv0, Wv1, bv1, Wv2, bv2,
                                  Wa0, ba0, Wa1, ba1, out, wsf);
  k2_wc<<<64, 256, 0, stream>>>(Wa0, Wa1, wc);
  k3_wmma<<<256, 32, 0, stream>>>(ope_gat, insert_gat, wc, wsf);
  k4_attn<<<dim3(4, 32, 8), 256, 0, stream>>>(wsf, Wa2, ba2, out);
}
